// Token_HyperlinkPredictionHead_72138270703714
// MI455X (gfx1250) — matrix operand order repaired and checked by reading, not compiled, not run on hardware
//
#include <hip/hip_runtime.h>
#include <hip/hip_bf16.h>

// ---------------------------------------------------------------------------
// Adaptive-softmax head for MI455X (gfx1250, wave32, WMMA bf16->f32)
//   - weights converted f32->bf16 once per call (reused by 128 row tiles)
//   - fused GEMM + per-lane online logsumexp (no shuffles in hot loop)
//   - branchless register capture of target / routing logits
// ---------------------------------------------------------------------------

typedef __attribute__((ext_vector_type(16))) __bf16 v16bf;
typedef __attribute__((ext_vector_type(8)))  float  v8f;

#define N_ROWS   2048      // B*K = 8*256
#define HDIM_    1024
#define CUT0     2000
#define CUT1     10000
#define C_HEAD   2002      // shortlist + 2 routing logits
#define C_OUT0   8000
#define C_OUT1   40257
#define H0_      256
#define H1_      64
#define NEG_BIG  (-3.0e38f)

__device__ __forceinline__ unsigned short bfbits(float f) {
    unsigned u = __builtin_bit_cast(unsigned, f);
    return (unsigned short)((u + 0x7FFFu + ((u >> 16) & 1u)) >> 16);  // RNE
}
__device__ __forceinline__ unsigned pack2(float a, float b) {
    return (unsigned)bfbits(a) | ((unsigned)bfbits(b) << 16);
}

// --------------------- bulk f32 -> bf16 weight conversion ------------------
__global__ __launch_bounds__(256)
void cvt_f32_bf16_kernel(const float* __restrict__ src, __bf16* __restrict__ dst, int n) {
    const int i = (blockIdx.x * 256 + threadIdx.x) * 8;
    if (i >= n) return;                       // n is a multiple of 8
    float4 a = *(const float4*)(src + i);
    float4 b = *(const float4*)(src + i + 4);
    uint4 p;
    p.x = pack2(a.x, a.y); p.y = pack2(a.z, a.w);
    p.z = pack2(b.x, b.y); p.w = pack2(b.z, b.w);
    *(uint4*)(dst + i) = p;
}

// -------------------------- gather + fp32->bf16 ---------------------------
__global__ __launch_bounds__(256)
void gather_cvt_kernel(const float* __restrict__ lhs, const int* __restrict__ idx,
                       __bf16* __restrict__ xbf, float* __restrict__ dout) {
    const int n   = blockIdx.x;               // row 0..2047
    const int tid = threadIdx.x;
    if (n == 0 && tid == 0) dout[N_ROWS] = 0.0f;   // zero loss accumulator
    if (tid >= 128) return;                   // 128 threads x 8 elems = 1024
    const int b = n >> 8;
    const int k = n & 255;
    const int s = idx[b * 256 + k];
    const float* src = lhs + ((size_t)b * 512 + (size_t)s) * HDIM_ + tid * 8;
    float4 a = *(const float4*)(src);
    float4 c = *(const float4*)(src + 4);
    uint4 p;
    p.x = pack2(a.x, a.y); p.y = pack2(a.z, a.w);
    p.z = pack2(c.x, c.y); p.w = pack2(c.z, c.w);
    *(uint4*)(xbf + (size_t)n * HDIM_ + tid * 8) = p;
}

// -------------------- WMMA fragment loaders (ISA layouts) ------------------
// A 16x32 bf16 (row-major in LDS): elems 0..7 -> K=8*half+e ; 8..15 -> +16
template<int KD>
__device__ __forceinline__ v16bf load_a_frag(const __bf16* sA, int r, int half, int k0) {
    v16bf a;
    const __bf16* ap = sA + r * KD + k0 + 8 * half;
    *(uint4*)&a       = *(const uint4*)ap;
    *((uint4*)&a + 1) = *(const uint4*)(ap + 16);
    return a;
}
// B 32x16 bf16 from bf16 W[C][KD] (col = r): elems 0..7 -> K=8*half+e ; 8..15 -> +16
template<int KD>
__device__ __forceinline__ v16bf load_b_frag(const __bf16* __restrict__ W,
                                             int colc, int half, int k0) {
    v16bf b;
    const __bf16* wp = W + (size_t)colc * KD + k0 + 8 * half;
    *(uint4*)&b       = *(const uint4*)wp;
    *((uint4*)&b + 1) = *(const uint4*)(wp + 16);
    return b;
}

// pairwise (max, sumexp) merge across the 16 lanes of a half-wave
__device__ __forceinline__ void merge16(float& m, float& s) {
#pragma unroll
    for (int off = 1; off < 16; off <<= 1) {
        float mo = __shfl_xor(m, off, 32);
        float so = __shfl_xor(s, off, 32);
        float mn = fmaxf(m, mo);
        s = s * __expf(m - mn) + so * __expf(mo - mn);
        m = mn;
    }
}
__device__ __forceinline__ float max16(float x) {
#pragma unroll
    for (int off = 1; off < 16; off <<= 1) x = fmaxf(x, __shfl_xor(x, off, 32));
    return x;
}

// --------------- fused GEMM + online logsumexp + target capture ------------
// A: bf16 [N_ROWS x KD], W: bf16 [C x KD].  Grid: (N_ROWS/16, nsplit), 256 thr.
template<int KD, bool HAS_BIAS, bool HAS_SPEC>
__global__ __launch_bounds__(256)
void fused_gemm_lse_kernel(const __bf16* __restrict__ A, const __bf16* __restrict__ W,
                           const float* __restrict__ bias,
                           int C, int clipMax, int tOff, int spec0, int spec1,
                           const int* __restrict__ targets,
                           float* __restrict__ pm, float* __restrict__ ps,
                           float* __restrict__ tcap,
                           float* __restrict__ cap0, float* __restrict__ cap1) {
    __shared__ __align__(16) __bf16 sA[16 * KD];
    const int tid  = threadIdx.x;
    const int wave = tid >> 5;
    const int lane = tid & 31;
    const int half = lane >> 4;
    const int r    = lane & 15;
    const int rowBase = blockIdx.x * 16;

    {   // stage 16 rows of A into LDS (contiguous, vectorized b128)
        const uint4* src = (const uint4*)(A + (size_t)rowBase * KD);
        uint4* dst = (uint4*)sA;
        for (int i = tid; i < 16 * KD / 8; i += 256) dst[i] = src[i];
    }
    __syncthreads();

    const int numTiles = (C + 15) >> 4;
    const int pstride  = gridDim.y * 8;
    const int pidx     = blockIdx.y * 8 + wave;

    int tclip[8];
#pragma unroll
    for (int v = 0; v < 8; ++v) {
        int tc = targets[rowBase + v + 8 * half] - tOff;
        tc = tc < 0 ? 0 : tc;
        tclip[v] = tc > clipMax ? clipMax : tc;
    }

    float rm[8], rs[8], capT[8], capS0[8], capS1[8];
#pragma unroll
    for (int v = 0; v < 8; ++v) {
        rm[v] = NEG_BIG; rs[v] = 0.0f;
        capT[v] = NEG_BIG; capS0[v] = NEG_BIG; capS1[v] = NEG_BIG;
    }

    for (int t = pidx; t < numTiles; t += pstride) {
        const int col   = (t << 4) + r;
        const bool valid = col < C;
        const int colc  = valid ? col : (C - 1);
        v8f acc = {};
        for (int k0 = 0; k0 < KD; k0 += 32) {
            v16bf a = load_a_frag<KD>(sA, r, half, k0);
            v16bf b = load_b_frag<KD>(W, colc, half, k0);
            acc = __builtin_amdgcn_wmma_f32_16x16x32_bf16(
                false, a, false, b, (short)0, acc, false, false);
        }
        float bc = 0.0f;
        if (HAS_BIAS) bc = bias[colc];
        const bool isS0 = HAS_SPEC && (col == spec0);
        const bool isS1 = HAS_SPEC && (col == spec1);
#pragma unroll
        for (int v = 0; v < 8; ++v) {
            const float c = valid ? (acc[v] + bc) : NEG_BIG;
            capT[v] = (valid && col == tclip[v]) ? c : capT[v];
            if (HAS_SPEC) {
                capS0[v] = isS0 ? c : capS0[v];
                capS1[v] = isS1 ? c : capS1[v];
            }
            // per-lane online logsumexp (no cross-lane traffic in hot loop)
            const float mn = fmaxf(rm[v], c);
            rs[v] = rs[v] * __expf(rm[v] - mn) + __expf(c - mn);
            rm[v] = mn;
        }
    }

    // cross-lane merges (once per kernel)
#pragma unroll
    for (int v = 0; v < 8; ++v) {
        merge16(rm[v], rs[v]);
        const float cT = max16(capT[v]);
        const int row  = rowBase + v + 8 * half;
        if (r == 0) {
            pm[(size_t)pidx * N_ROWS + row] = rm[v];
            ps[(size_t)pidx * N_ROWS + row] = rs[v];
            if (cT > -1.0e37f) tcap[row] = cT;
        }
        if (HAS_SPEC) {
            const float c0 = max16(capS0[v]);
            const float c1 = max16(capS1[v]);
            if (r == 0) {
                if (c0 > -1.0e37f) cap0[row] = c0;
                if (c1 > -1.0e37f) cap1[row] = c1;
            }
        }
    }
}

// ------------------- plain WMMA GEMM -> bf16 output (hidden) ---------------
template<int KD>
__global__ __launch_bounds__(256)
void gemm_bf16_kernel(const __bf16* __restrict__ A, const __bf16* __restrict__ W,
                      __bf16* __restrict__ out, int C) {
    __shared__ __align__(16) __bf16 sA[16 * KD];
    const int tid  = threadIdx.x;
    const int wave = tid >> 5;
    const int lane = tid & 31;
    const int half = lane >> 4;
    const int r    = lane & 15;
    const int rowBase = blockIdx.x * 16;
    {
        const uint4* src = (const uint4*)(A + (size_t)rowBase * KD);
        uint4* dst = (uint4*)sA;
        for (int i = tid; i < 16 * KD / 8; i += 256) dst[i] = src[i];
    }
    __syncthreads();

    const int colBase = (blockIdx.y * 8 + wave) * 16;
    if (colBase >= C) return;
    const int col = colBase + r;
    v8f acc = {};
    for (int k0 = 0; k0 < KD; k0 += 32) {
        v16bf a = load_a_frag<KD>(sA, r, half, k0);
        v16bf b = load_b_frag<KD>(W, col, half, k0);
        acc = __builtin_amdgcn_wmma_f32_16x16x32_bf16(
            false, a, false, b, (short)0, acc, false, false);
    }
#pragma unroll
    for (int v = 0; v < 8; ++v) {
        const int row = rowBase + v + 8 * half;
        out[(size_t)row * C + col] = __builtin_bit_cast(__bf16, bfbits(acc[v]));
    }
}

// ---------------------- merge partials + final selection -------------------
__device__ __forceinline__ float lse_merge(const float* pm, const float* ps,
                                           int P, int n) {
    float m = NEG_BIG, s = 0.0f;
    for (int i = 0; i < P; ++i) {
        const float mi = pm[(size_t)i * N_ROWS + n];
        const float si = ps[(size_t)i * N_ROWS + n];
        const float mn = fmaxf(m, mi);
        s = s * __expf(m - mn) + si * __expf(mi - mn);
        m = mn;
    }
    return m + __logf(s);
}

__global__ __launch_bounds__(256)
void combine_kernel(const int* __restrict__ targets,
                    const float* pmH, const float* psH,
                    const float* pm0, const float* ps0,
                    const float* pm1, const float* ps1,
                    const float* tcapH, const float* capC0, const float* capC1,
                    const float* tcap0, const float* tcap1,
                    float* __restrict__ out) {
    const int n = blockIdx.x * 256 + threadIdx.x;
    const float lseH = lse_merge(pmH, psH, 32, n);
    const float lse0 = lse_merge(pm0, ps0, 64, n);
    const float lse1 = lse_merge(pm1, ps1, 128, n);
    const int   t    = targets[n];
    const float sv   = tcapH[n] - lseH;
    const float mid  = (capC0[n] - lseH) + (tcap0[n] - lse0);
    const float high = (capC1[n] - lseH) + (tcap1[n] - lse1);
    const float o = (t < CUT0) ? sv : ((t < CUT1) ? mid : high);
    out[n] = o;

    __shared__ float red[256];
    red[threadIdx.x] = o;
    __syncthreads();
#pragma unroll
    for (int s = 128; s > 0; s >>= 1) {
        if (threadIdx.x < s) red[threadIdx.x] += red[threadIdx.x + s];
        __syncthreads();
    }
    if (threadIdx.x == 0)
        atomicAdd(out + N_ROWS, -red[0] * (1.0f / (float)N_ROWS));
}

// ------------------------------- launcher ----------------------------------
extern "C" void kernel_launch(void* const* d_in, const int* in_sizes, int n_in,
                              void* d_out, int out_size, void* d_ws, size_t ws_size,
                              hipStream_t stream) {
    (void)in_sizes; (void)n_in; (void)out_size; (void)ws_size;
    const int*   targets = (const int*)  d_in[0];
    const float* lhs     = (const float*)d_in[1];
    // d_in[2] = cls_tokens_batch (unused by reference)
    const int*   idx     = (const int*)  d_in[3];
    const float* headW   = (const float*)d_in[4];
    const float* headB   = (const float*)d_in[5];
    const float* proj0W  = (const float*)d_in[6];
    const float* out0W   = (const float*)d_in[7];
    const float* proj1W  = (const float*)d_in[8];
    const float* out1W   = (const float*)d_in[9];

    char* ws = (char*)d_ws;
    size_t off = 0;
    __bf16* xbf  = (__bf16*)(ws + off); off += (size_t)N_ROWS * HDIM_ * 2;
    __bf16* hid0 = (__bf16*)(ws + off); off += (size_t)N_ROWS * H0_  * 2;
    __bf16* hid1 = (__bf16*)(ws + off); off += (size_t)N_ROWS * H1_  * 2;
    __bf16* headWb  = (__bf16*)(ws + off); off += (size_t)C_HEAD * HDIM_ * 2;
    __bf16* proj0Wb = (__bf16*)(ws + off); off += (size_t)H0_ * HDIM_ * 2;
    __bf16* proj1Wb = (__bf16*)(ws + off); off += (size_t)H1_ * HDIM_ * 2;
    __bf16* out0Wb  = (__bf16*)(ws + off); off += (size_t)C_OUT0 * H0_ * 2;
    __bf16* out1Wb  = (__bf16*)(ws + off); off += (size_t)C_OUT1 * H1_ * 2;
    float* pmH = (float*)(ws + off); off += (size_t)32  * N_ROWS * 4;
    float* psH = (float*)(ws + off); off += (size_t)32  * N_ROWS * 4;
    float* pm0 = (float*)(ws + off); off += (size_t)64  * N_ROWS * 4;
    float* ps0 = (float*)(ws + off); off += (size_t)64  * N_ROWS * 4;
    float* pm1 = (float*)(ws + off); off += (size_t)128 * N_ROWS * 4;
    float* ps1 = (float*)(ws + off); off += (size_t)128 * N_ROWS * 4;
    float* tH  = (float*)(ws + off); off += (size_t)N_ROWS * 4;
    float* hc0 = (float*)(ws + off); off += (size_t)N_ROWS * 4;
    float* hc1 = (float*)(ws + off); off += (size_t)N_ROWS * 4;
    float* t0c = (float*)(ws + off); off += (size_t)N_ROWS * 4;
    float* t1c = (float*)(ws + off); off += (size_t)N_ROWS * 4;

    float* out = (float*)d_out;   // [0..2047]=target logprobs, [2048]=loss

    // 0) one-time (per call) weight conversions to bf16
    auto cvt = [&](const float* s, __bf16* d, int n) {
        cvt_f32_bf16_kernel<<<(n / 8 + 255) / 256, 256, 0, stream>>>(s, d, n);
    };
    cvt(headW,  headWb,  C_HEAD * HDIM_);
    cvt(proj0W, proj0Wb, H0_ * HDIM_);
    cvt(proj1W, proj1Wb, H1_ * HDIM_);
    cvt(out0W,  out0Wb,  C_OUT0 * H0_);
    cvt(out1W,  out1Wb,  C_OUT1 * H1_);

    // 1) gather + convert to bf16 (also zeroes loss slot)
    gather_cvt_kernel<<<N_ROWS, 256, 0, stream>>>(lhs, idx, xbf, out);

    // 2) head: x @ head_W.T + b, fused online logsumexp + routing capture
    fused_gemm_lse_kernel<HDIM_, true, true><<<dim3(N_ROWS / 16, 4), 256, 0, stream>>>(
        xbf, headWb, headB, C_HEAD, CUT0 - 1, 0, CUT0, CUT0 + 1,
        targets, pmH, psH, tH, hc0, hc1);

    // 3) low-rank projections -> bf16 hiddens
    gemm_bf16_kernel<HDIM_><<<dim3(N_ROWS / 16, 2), 256, 0, stream>>>(xbf, proj0Wb, hid0, H0_);
    gemm_bf16_kernel<HDIM_><<<dim3(N_ROWS / 16, 1), 256, 0, stream>>>(xbf, proj1Wb, hid1, H1_);

    // 4) cluster output GEMMs, fused logsumexp + target capture
    fused_gemm_lse_kernel<H0_, false, false><<<dim3(N_ROWS / 16, 8), 256, 0, stream>>>(
        hid0, out0Wb, nullptr, C_OUT0, C_OUT0 - 1, CUT0, -1, -1,
        targets, pm0, ps0, t0c, t0c, t0c);
    fused_gemm_lse_kernel<H1_, false, false><<<dim3(N_ROWS / 16, 16), 256, 0, stream>>>(
        hid1, out1Wb, nullptr, C_OUT1, C_OUT1 - 1, CUT1, -1, -1,
        targets, pm1, ps1, t1c, t1c, t1c);

    // 5) merge partial logsumexps, branch-select, loss
    combine_kernel<<<N_ROWS / 256, 256, 0, stream>>>(
        targets, pmH, psH, pm0, ps0, pm1, ps1, tH, hc0, hc1, t0c, t1c, out);
}
